// GNN_76862734730017
// MI455X (gfx1250) — compile-verified, matrix operand order checked by reading, not run on hardware
//
#include <hip/hip_runtime.h>
#include <hip/hip_bf16.h>

typedef __attribute__((ext_vector_type(16))) __bf16 v16bf;
typedef __attribute__((ext_vector_type(8)))  float  v8f;

#define IN_FEAT 512
#define HID     16

__device__ __forceinline__ void atomAddF(float* p, float v) {
    // relaxed agent-scope fp32 add -> global_atomic_add_f32 on gfx1250
    __hip_atomic_fetch_add(p, v, __ATOMIC_RELAXED, __HIP_MEMORY_SCOPE_AGENT);
}

// ---------------- degree / normalization ----------------
__global__ void deg_init(float* deg, int N) {
    int i = blockIdx.x * blockDim.x + threadIdx.x;
    if (i < N) deg[i] = 1.0f;                       // self-loop
}

__global__ void deg_count(const int* __restrict__ dst, float* deg, int E) {
    int e = blockIdx.x * blockDim.x + threadIdx.x;
    if (e < E) atomAddF(&deg[dst[e]], 1.0f);
}

__global__ void deg_rsqrt(float* deg, int N) {
    int i = blockIdx.x * blockDim.x + threadIdx.x;
    if (i < N) {
        float d = deg[i];
        deg[i] = (d > 0.0f) ? rsqrtf(d) : 0.0f;     // now holds dinv
    }
}

// ---------------- GEMM1: h1 = x @ W1 via WMMA bf16 ----------------
__global__ void __launch_bounds__(256)
gemm1_wmma(const float* __restrict__ x, const float* __restrict__ W1,
           float* __restrict__ h1, int nRows) {
    // B fragments for all 16 k-tiles, WMMA lane order, bf16: 16KB LDS
    __shared__ __bf16 Blds[16 * 32 * 16];

    // cooperative fill: idx = kt*512 + lane*16 + e
    for (int idx = threadIdx.x; idx < 16 * 512; idx += blockDim.x) {
        int kt   = idx >> 9;
        int rem  = idx & 511;
        int lane = rem >> 4;
        int e    = rem & 15;
        int col  = lane & 15;
        int kb   = (lane >> 4) * 8;
        int k    = kt * 32 + ((e >> 3) ? 16 : 0) + kb + (e & 7);
        Blds[idx] = (__bf16)W1[k * HID + col];
    }
    __syncthreads();

    int wave = (blockIdx.x * blockDim.x + threadIdx.x) >> 5;
    int lane = threadIdx.x & 31;
    int r0   = wave * 16;
    if (r0 >= nRows) return;                        // wave-uniform exit

    int row = r0 + (lane & 15);
    int kb  = (lane >> 4) * 8;                      // K sub-offset for this lane half
    const float* xr = x + (long)row * IN_FEAT;

    v8f acc = {};
    #pragma unroll
    for (int kt = 0; kt < 16; ++kt) {
        int k0 = kt * 32 + kb;
        float4 f0 = *(const float4*)(xr + k0);
        float4 f1 = *(const float4*)(xr + k0 + 4);
        float4 f2 = *(const float4*)(xr + k0 + 16);
        float4 f3 = *(const float4*)(xr + k0 + 20);
        v16bf a;
        a[0]=(__bf16)f0.x;  a[1]=(__bf16)f0.y;  a[2]=(__bf16)f0.z;  a[3]=(__bf16)f0.w;
        a[4]=(__bf16)f1.x;  a[5]=(__bf16)f1.y;  a[6]=(__bf16)f1.z;  a[7]=(__bf16)f1.w;
        a[8]=(__bf16)f2.x;  a[9]=(__bf16)f2.y;  a[10]=(__bf16)f2.z; a[11]=(__bf16)f2.w;
        a[12]=(__bf16)f3.x; a[13]=(__bf16)f3.y; a[14]=(__bf16)f3.z; a[15]=(__bf16)f3.w;

        v16bf b = *(const v16bf*)&Blds[kt * 512 + lane * 16];
        acc = __builtin_amdgcn_wmma_f32_16x16x32_bf16(
                  false, a, false, b, (short)0, acc, false, false);
    }

    // D layout: VGPR v, lanes 0-15 -> M=v, lanes 16-31 -> M=v+8; N = lane%16
    int outRow0 = r0 + (lane >> 4) * 8;
    int col     = lane & 15;
    #pragma unroll
    for (int v = 0; v < 8; ++v)
        h1[(long)(outRow0 + v) * HID + col] = acc[v];
}

// ---------------- aggregation (16 features) ----------------
__global__ void agg_init16(const float* __restrict__ h, const float* __restrict__ dinv,
                           float* __restrict__ agg, int N) {
    int t = blockIdx.x * blockDim.x + threadIdx.x;   // one float4 per thread
    if (t >= N * 4) return;
    int i = t >> 2;
    float w = dinv[i] * dinv[i];
    float4 v = ((const float4*)h)[t];
    float4 r; r.x = v.x * w; r.y = v.y * w; r.z = v.z * w; r.w = v.w * w;
    ((float4*)agg)[t] = r;
}

__global__ void agg_edges16(const int* __restrict__ src, const int* __restrict__ dst,
                            const float* __restrict__ dinv, const float* __restrict__ h,
                            float* __restrict__ agg, int E) {
    int e = blockIdx.x * blockDim.x + threadIdx.x;
    if (e >= E) return;
    int s = src[e], d = dst[e];
    float w = dinv[s] * dinv[d];
    const float4* hp = (const float4*)(h + (long)s * 16);
    float4 a = hp[0], b = hp[1], c = hp[2], q = hp[3];
    float* ap = agg + (long)d * 16;
    atomAddF(ap + 0,  a.x * w); atomAddF(ap + 1,  a.y * w);
    atomAddF(ap + 2,  a.z * w); atomAddF(ap + 3,  a.w * w);
    atomAddF(ap + 4,  b.x * w); atomAddF(ap + 5,  b.y * w);
    atomAddF(ap + 6,  b.z * w); atomAddF(ap + 7,  b.w * w);
    atomAddF(ap + 8,  c.x * w); atomAddF(ap + 9,  c.y * w);
    atomAddF(ap + 10, c.z * w); atomAddF(ap + 11, c.w * w);
    atomAddF(ap + 12, q.x * w); atomAddF(ap + 13, q.y * w);
    atomAddF(ap + 14, q.z * w); atomAddF(ap + 15, q.w * w);
}

// ---------------- layer 2: h2 = relu(agg1 + b1) @ W2 (pad N to 8) ----------------
__global__ void layer2_kernel(const float* __restrict__ agg1, const float* __restrict__ b1,
                              const float* __restrict__ W2, float* __restrict__ h2, int N) {
    int i = blockIdx.x * blockDim.x + threadIdx.x;
    if (i >= N) return;
    float h[16];
    const float4* ap = (const float4*)(agg1 + (long)i * 16);
    #pragma unroll
    for (int q = 0; q < 4; ++q) {
        float4 v = ap[q];
        h[q*4+0] = fmaxf(v.x + b1[q*4+0], 0.0f);
        h[q*4+1] = fmaxf(v.y + b1[q*4+1], 0.0f);
        h[q*4+2] = fmaxf(v.z + b1[q*4+2], 0.0f);
        h[q*4+3] = fmaxf(v.w + b1[q*4+3], 0.0f);
    }
    float* op = h2 + (long)i * 8;
    #pragma unroll
    for (int n = 0; n < 7; ++n) {
        float s = 0.0f;
        #pragma unroll
        for (int k = 0; k < 16; ++k) s += h[k] * W2[k * 7 + n];
        op[n] = s;
    }
    op[7] = 0.0f;
}

// ---------------- aggregation (8-wide rows, 7 used) ----------------
__global__ void agg_init8(const float* __restrict__ h, const float* __restrict__ dinv,
                          float* __restrict__ agg, int N) {
    int t = blockIdx.x * blockDim.x + threadIdx.x;   // one float4 per thread
    if (t >= N * 2) return;
    int i = t >> 1;
    float w = dinv[i] * dinv[i];
    float4 v = ((const float4*)h)[t];
    float4 r; r.x = v.x * w; r.y = v.y * w; r.z = v.z * w; r.w = v.w * w;
    ((float4*)agg)[t] = r;
}

__global__ void agg_edges8(const int* __restrict__ src, const int* __restrict__ dst,
                           const float* __restrict__ dinv, const float* __restrict__ h,
                           float* __restrict__ agg, int E) {
    int e = blockIdx.x * blockDim.x + threadIdx.x;
    if (e >= E) return;
    int s = src[e], d = dst[e];
    float w = dinv[s] * dinv[d];
    const float4* hp = (const float4*)(h + (long)s * 8);
    float4 a = hp[0], b = hp[1];
    float* ap = agg + (long)d * 8;
    atomAddF(ap + 0, a.x * w); atomAddF(ap + 1, a.y * w);
    atomAddF(ap + 2, a.z * w); atomAddF(ap + 3, a.w * w);
    atomAddF(ap + 4, b.x * w); atomAddF(ap + 5, b.y * w);
    atomAddF(ap + 6, b.z * w);
}

// ---------------- bias + log_softmax ----------------
__global__ void logsoftmax_kernel(const float* __restrict__ agg2, const float* __restrict__ b2,
                                  float* __restrict__ out, int N) {
    int i = blockIdx.x * blockDim.x + threadIdx.x;
    if (i >= N) return;
    float v[7];
    const float* ap = agg2 + (long)i * 8;
    #pragma unroll
    for (int n = 0; n < 7; ++n) v[n] = ap[n] + b2[n];
    float m = v[0];
    #pragma unroll
    for (int n = 1; n < 7; ++n) m = fmaxf(m, v[n]);
    float s = 0.0f;
    #pragma unroll
    for (int n = 0; n < 7; ++n) s += __expf(v[n] - m);
    float l = __logf(s);
    float* op = out + (long)i * 7;
    #pragma unroll
    for (int n = 0; n < 7; ++n) op[n] = v[n] - m - l;
}

extern "C" void kernel_launch(void* const* d_in, const int* in_sizes, int n_in,
                              void* d_out, int out_size, void* d_ws, size_t ws_size,
                              hipStream_t stream) {
    const float* x  = (const float*)d_in[0];
    const int*   ei = (const int*)  d_in[1];
    const float* W1 = (const float*)d_in[2];
    const float* b1 = (const float*)d_in[3];
    const float* W2 = (const float*)d_in[4];
    const float* b2 = (const float*)d_in[5];

    const int N = in_sizes[0] / IN_FEAT;   // 100000
    const int E = in_sizes[1] / 2;         // 3200000
    const int* src = ei;
    const int* dst = ei + E;

    float* ws   = (float*)d_ws;
    float* dinv = ws;                         // N
    float* h1   = dinv + (size_t)N;           // N*16
    float* agg1 = h1   + (size_t)N * 16;      // N*16
    float* h2   = agg1 + (size_t)N * 16;      // N*8
    float* agg2 = h2   + (size_t)N * 8;       // N*8

    const int B = 256;
    auto cdiv = [](long a, long b) { return (int)((a + b - 1) / b); };

    // degrees -> dinv
    deg_init <<<cdiv(N, B), B, 0, stream>>>(dinv, N);
    deg_count<<<cdiv(E, B), B, 0, stream>>>(dst, dinv, E);
    deg_rsqrt<<<cdiv(N, B), B, 0, stream>>>(dinv, N);

    // layer 1
    {
        int waves  = (N + 15) / 16;           // 6250
        int blocks = (waves + 7) / 8;         // 8 waves / 256-thread block
        gemm1_wmma<<<blocks, B, 0, stream>>>(x, W1, h1, N);
    }
    agg_init16 <<<cdiv((long)N * 4, B), B, 0, stream>>>(h1, dinv, agg1, N);
    agg_edges16<<<cdiv(E, B), B, 0, stream>>>(src, dst, dinv, h1, agg1, E);

    // layer 2
    layer2_kernel<<<cdiv(N, B), B, 0, stream>>>(agg1, b1, W2, h2, N);
    agg_init8 <<<cdiv((long)N * 2, B), B, 0, stream>>>(h2, dinv, agg2, N);
    agg_edges8<<<cdiv(E, B), B, 0, stream>>>(src, dst, dinv, h2, agg2, E);

    // output
    logsoftmax_kernel<<<cdiv(N, B), B, 0, stream>>>(agg2, b2, (float*)d_out, N);
}
